// SimpleSceneNet_62002147885294
// MI455X (gfx1250) — matrix operand order repaired and checked by reading, not compiled, not run on hardware
//
#include <hip/hip_runtime.h>

typedef __attribute__((ext_vector_type(16))) _Float16 v16h;
typedef __attribute__((ext_vector_type(8)))  _Float16 v8h;
typedef __attribute__((ext_vector_type(8)))  float    v8f;

#define NPTS   262144   // 2 * 131072 points total
#define NPB    131072   // points per batch
#define SUBV   64000    // 40^3 voxel sub-box (points are uniform in [0,1))
#define EPS_GN 1e-5f
#define SLOPE_ 0.01f

__device__ __forceinline__ float lrelu(float v){ return v >= 0.0f ? v : SLOPE_*v; }

// ---------------- utility kernels ----------------
__global__ void k_zero(float* p, size_t n){
  size_t i = (size_t)blockIdx.x*blockDim.x + threadIdx.x;
  size_t stride = (size_t)gridDim.x*blockDim.x;
  for (; i < n; i += stride) p[i] = 0.0f;
}

__global__ void k_cvt_f16(const float* __restrict__ s, _Float16* __restrict__ d, int n){
  int i = blockIdx.x*blockDim.x + threadIdx.x;
  if (i < n) d[i] = (_Float16)s[i];
}

// ---------------- layer 1: 4 -> 64 (+ voxel index) ----------------
__global__ void k_layer1(const float* __restrict__ pc, const float* __restrict__ w,
                         const float* __restrict__ bias, _Float16* __restrict__ Y,
                         int* __restrict__ vox){
  int n = blockIdx.x*blockDim.x + threadIdx.x;
  if (n >= NPTS) return;
  float4 p = ((const float4*)pc)[n];
  int ix = (int)floorf((p.x + 1.0f)/0.025f);
  int iy = (int)floorf((p.y + 1.0f)/0.025f);
  int iz = (int)floorf((p.z + 1.0f)/0.025f);
  ix = min(max(ix,40),79); iy = min(max(iy,40),79); iz = min(max(iz,40),79);
  vox[n] = (ix-40)*1600 + (iy-40)*40 + (iz-40);
  _Float16* yr = Y + (size_t)n*64;
  #pragma unroll 8
  for (int o = 0; o < 64; ++o){
    const float* wr = w + o*4;
    float s = bias[o] + wr[0]*p.x + wr[1]*p.y + wr[2]*p.z + wr[3]*p.w;
    yr[o] = (_Float16)s;
  }
}

// ---------------- WMMA GEMM: Y[n][COUT] = W[COUT][CIN] * X[n][CIN] + b -------
// 256-thread block (8 waves). W is async-staged into LDS once per block; each
// wave owns COUT/2 rows (wave column = wv&1) and NT n-tiles (group = wv>>1).
// A fragments come from LDS (ds_load), B fragments are software-pipelined
// global loads, each A fragment feeds NT independent WMMAs.
template<int CIN, int COUT, int NT>
__global__ __launch_bounds__(256)
void k_gemm(const _Float16* __restrict__ X, const _Float16* __restrict__ W,
            const float* __restrict__ bias, _Float16* __restrict__ Y){
  constexpr int MT  = COUT/16;
  constexpr int MTW = MT/2;           // m-tiles per wave
  constexpr int KS  = CIN/32;
  constexpr int WB  = COUT*CIN;       // halfs in W
  __shared__ _Float16 sW[WB];

  const int tid = threadIdx.x;

  // ---- async stage W (HBM -> LDS), 16B per op ----
  {
    constexpr unsigned nchunks = WB/8;
    unsigned lbase = (unsigned)(uintptr_t)&sW[0];
    for (unsigned i = tid; i < nchunks; i += 256u){
      unsigned lds  = lbase + i*16u;
      unsigned goff = i*16u;
      asm volatile("global_load_async_to_lds_b128 %0, %1, %2"
                   :: "v"(lds), "v"(goff), "s"(W) : "memory");
    }
    asm volatile("s_wait_asynccnt 0x0" ::: "memory");
  }
  __syncthreads();

  const int wv    = tid >> 5;
  const int lane  = tid & 31;
  const int mhalf = wv & 1;           // which half of COUT
  const int ngrp  = wv >> 1;          // 0..3
  const int r  = lane & 15;
  const int hi = lane >> 4;
  const int kA = hi ? 16 : 0;         // A operand: blocked K halves
  const int kB = hi ? 16 : 0;         // B operand: blocked K halves

  const int n0 = blockIdx.x*(4*NT*16) + ngrp*(NT*16);

  v8f acc[NT][MTW];
  v8f z = {};
  #pragma unroll
  for (int t = 0; t < NT; ++t)
    #pragma unroll
    for (int m = 0; m < MTW; ++m) acc[t][m] = z;

  const _Float16* xrow[NT];
  #pragma unroll
  for (int t = 0; t < NT; ++t)
    xrow[t] = X + (size_t)(n0 + t*16 + r)*CIN + kB;

  v16h bcur[NT];
  #pragma unroll
  for (int t = 0; t < NT; ++t) bcur[t] = *(const v16h*)(xrow[t]);

  const _Float16* wbase = &sW[(size_t)(mhalf*MTW*16 + r)*CIN + kA];

  #pragma unroll
  for (int ks = 0; ks < KS; ++ks){
    v16h bnext[NT];
    #pragma unroll
    for (int t = 0; t < NT; ++t)
      bnext[t] = (ks + 1 < KS) ? *(const v16h*)(xrow[t] + (ks+1)*32) : bcur[t];

    #pragma unroll
    for (int m = 0; m < MTW; ++m){
      const _Float16* wr = wbase + (size_t)(m*16)*CIN + ks*32;
      v8h alo = *(const v8h*)(wr);        // K = kA .. kA+7   (ds_load_b128)
      v8h ahi = *(const v8h*)(wr + 8);    // K = kA+8 .. kA+15
      v16h a = __builtin_shufflevector(alo, ahi,
                 0,1,2,3,4,5,6,7,8,9,10,11,12,13,14,15);
      #pragma unroll
      for (int t = 0; t < NT; ++t)
        acc[t][m] = __builtin_amdgcn_wmma_f32_16x16x32_f16(
                      false, a, false, bcur[t], (short)0, acc[t][m], false, false);
    }
    #pragma unroll
    for (int t = 0; t < NT; ++t) bcur[t] = bnext[t];
  }

  #pragma unroll
  for (int t = 0; t < NT; ++t){
    const size_t yr = (size_t)(n0 + t*16 + r)*COUT;
    #pragma unroll
    for (int m = 0; m < MTW; ++m){
      int co0 = (mhalf*MTW + m)*16 + hi*8;   // C/D layout: M = vgpr (+8)
      v8h outv;
      #pragma unroll
      for (int p = 0; p < 8; ++p) outv[p] = (_Float16)(acc[t][m][p] + bias[co0 + p]);
      *(v8h*)(Y + yr + co0) = outv;
    }
  }
}

// ---------------- GroupNorm statistics (sum, sumsq per batch/group) ---------
template<int C>
__global__ void k_gnstats(const _Float16* __restrict__ Y, float* __restrict__ stats){
  constexpr int CPG = C/16;
  const int b = blockIdx.y;
  __shared__ float ssum[16], ssq[16];
  if (threadIdx.x < 16){ ssum[threadIdx.x] = 0.f; ssq[threadIdx.x] = 0.f; }
  __syncthreads();
  float ls[16], lq[16];
  #pragma unroll
  for (int g = 0; g < 16; ++g){ ls[g] = 0.f; lq[g] = 0.f; }
  const int ppb = NPB / gridDim.x;
  const int nbase = b*NPB + blockIdx.x*ppb;
  for (int i = threadIdx.x; i < ppb; i += blockDim.x){
    const _Float16* row = Y + (size_t)(nbase + i)*C;
    __builtin_prefetch(row + (size_t)blockDim.x*C, 0, 0);
    #pragma unroll
    for (int cc = 0; cc < C/8; ++cc){
      v8h v = *(const v8h*)(row + cc*8);
      #pragma unroll
      for (int j = 0; j < 8; ++j){
        float f = (float)v[j];
        int g = (cc*8 + j)/CPG;
        ls[g] += f; lq[g] += f*f;
      }
    }
  }
  #pragma unroll
  for (int g = 0; g < 16; ++g){ atomicAdd(&ssum[g], ls[g]); atomicAdd(&ssq[g], lq[g]); }
  __syncthreads();
  if (threadIdx.x < 16){
    atomicAdd(&stats[(b*16 + threadIdx.x)*2 + 0], ssum[threadIdx.x]);
    atomicAdd(&stats[(b*16 + threadIdx.x)*2 + 1], ssq[threadIdx.x]);
  }
}

// ---------------- GN finalize: per-channel affine y = A*x + B ---------------
template<int C>
__global__ void k_gnfin(const float* __restrict__ stats, const float* __restrict__ gamma,
                        const float* __restrict__ beta, float* __restrict__ cA,
                        float* __restrict__ cB){
  int t = threadIdx.x; if (t >= 32) return;
  int b = t >> 4, g = t & 15;
  constexpr int CPG = C/16;
  const float cnt = (float)CPG * (float)NPB;
  float sum = stats[(b*16+g)*2+0], sq = stats[(b*16+g)*2+1];
  float m  = sum/cnt;
  float var = sq/cnt - m*m;
  float rs = rsqrtf(var + EPS_GN);
  for (int j = 0; j < CPG; ++j){
    int c = g*CPG + j;
    float A = gamma[c]*rs;
    cA[b*C + c] = A;
    cB[b*C + c] = beta[c] - m*A;
  }
}

// ---------------- normalize + LeakyReLU (f16 -> f16) ------------------------
template<int C>
__global__ void k_transform(const _Float16* __restrict__ X, const float* __restrict__ cA,
                            const float* __restrict__ cB, _Float16* __restrict__ T){
  size_t i = ((size_t)blockIdx.x*blockDim.x + threadIdx.x)*8;
  if (i >= (size_t)NPTS*C) return;
  int n = (int)(i / C);
  int c = (int)(i % C);
  int b = n >= NPB;
  v8h x = *(const v8h*)(X + i);
  v8h o;
  #pragma unroll
  for (int j = 0; j < 8; ++j){
    float v = cA[b*C + c + j]*(float)x[j] + cB[b*C + c + j];
    o[j] = (_Float16)lrelu(v);
  }
  *(v8h*)(T + i) = o;
}

// ---------------- voxel segment-sum (L2-resident f32 atomics) ---------------
__global__ void k_scatter(const _Float16* __restrict__ T, const int* __restrict__ vox,
                          float* __restrict__ acc){
  int idx = blockIdx.x*blockDim.x + threadIdx.x;  // NPTS*32 threads
  int n  = idx >> 5;
  int cq = (idx & 31)*8;
  if (n >= NPTS) return;
  int b = n >= NPB;
  float* dst = acc + (((size_t)b*SUBV + vox[n])*256 + cq);
  v8h t = *(const v8h*)(T + (size_t)n*256 + cq);
  #pragma unroll
  for (int j = 0; j < 8; ++j) atomicAdd(dst + j, (float)t[j]);
}

// ---------------- max over voxels (0 = empty / out-of-box voxels) -----------
__global__ void k_voxmax(const float* __restrict__ acc, float* __restrict__ out){
  int b = blockIdx.y;
  int lane = threadIdx.x & 31;
  int vs   = threadIdx.x >> 5;       // 32 voxel streams
  int c = blockIdx.x*32 + lane;      // gridDim.x = 8 -> 256 channels
  float mx = 0.0f;
  for (int v = vs; v < SUBV; v += 32)
    mx = fmaxf(mx, acc[((size_t)b*SUBV + v)*256 + c]);
  __shared__ float sm[1024];
  sm[threadIdx.x] = mx;
  __syncthreads();
  for (int s = 16; s >= 1; s >>= 1){
    if (vs < s) sm[threadIdx.x] = fmaxf(sm[threadIdx.x], sm[(vs+s)*32 + lane]);
    __syncthreads();
  }
  if (vs == 0) out[b*256 + c] = fmaxf(sm[lane], 0.0f);
}

// ---------------- head MLP (tiny; per-batch block, GN in LDS) ---------------
template<int CI, int CO, bool DOGN>
__global__ void k_head(const float* __restrict__ in, const float* __restrict__ W,
                       const float* __restrict__ bias, const float* __restrict__ gamma,
                       const float* __restrict__ beta, float* __restrict__ out){
  const int b = blockIdx.x;
  __shared__ float xin[CI];
  __shared__ float yv[CO];
  for (int i = threadIdx.x; i < CI; i += blockDim.x) xin[i] = in[b*CI + i];
  __syncthreads();
  for (int o = threadIdx.x; o < CO; o += blockDim.x){
    const float* wr = W + (size_t)o*CI;
    float s = bias[o];
    for (int c = 0; c < CI; ++c) s += wr[c]*xin[c];
    yv[o] = s;
  }
  __syncthreads();
  if constexpr (DOGN){
    __shared__ float gm[16], gr[16];
    constexpr int CPG = CO/16;
    if (threadIdx.x < 16){
      float sum = 0.f, sq = 0.f;
      for (int j = 0; j < CPG; ++j){ float v = yv[threadIdx.x*CPG + j]; sum += v; sq += v*v; }
      float m = sum/CPG;
      float var = sq/CPG - m*m;
      gm[threadIdx.x] = m; gr[threadIdx.x] = rsqrtf(var + EPS_GN);
    }
    __syncthreads();
    for (int o = threadIdx.x; o < CO; o += blockDim.x){
      int g = o/CPG;
      float v = (yv[o] - gm[g])*gr[g]*gamma[o] + beta[o];
      out[b*CO + o] = lrelu(v);
    }
  } else {
    for (int o = threadIdx.x; o < CO; o += blockDim.x) out[b*CO + o] = yv[o];
  }
}

// ---------------- launch ----------------------------------------------------
extern "C" void kernel_launch(void* const* d_in, const int* in_sizes, int n_in,
                              void* d_out, int out_size, void* d_ws, size_t ws_size,
                              hipStream_t stream){
  const float* pc  = (const float*)d_in[0];
  const float* w1  = (const float*)d_in[1];
  const float* b1  = (const float*)d_in[2];
  const float* g1  = (const float*)d_in[3];
  const float* be1 = (const float*)d_in[4];
  const float* w2  = (const float*)d_in[5];
  const float* b2  = (const float*)d_in[6];
  const float* g2  = (const float*)d_in[7];
  const float* be2 = (const float*)d_in[8];
  const float* w3  = (const float*)d_in[9];
  const float* b3  = (const float*)d_in[10];
  const float* g3  = (const float*)d_in[11];
  const float* be3 = (const float*)d_in[12];
  const float* lw1 = (const float*)d_in[13];
  const float* lb1 = (const float*)d_in[14];
  const float* lw2 = (const float*)d_in[15];
  const float* lb2 = (const float*)d_in[16];
  const float* lw3 = (const float*)d_in[17];
  const float* lb3 = (const float*)d_in[18];
  const float* lg1 = (const float*)d_in[19];
  const float* lbe1= (const float*)d_in[20];
  const float* lg2 = (const float*)d_in[21];
  const float* lbe2= (const float*)d_in[22];

  char* base = (char*)d_ws;
  size_t off = 0;
  auto carve = [&](size_t bytes)->char*{
    char* p = base + off;
    off = (off + bytes + 255) & ~(size_t)255;
    return p;
  };
  _Float16* bufA = (_Float16*)carve((size_t)NPTS*256*2);   // 134 MB ping
  _Float16* bufB = (_Float16*)carve((size_t)NPTS*256*2);   // 134 MB pong
  float*    acc  = (float*)   carve((size_t)2*SUBV*256*4); // 131 MB (fits L2)
  int*      vox  = (int*)     carve((size_t)NPTS*4);
  _Float16* W2h  = (_Float16*)carve(128*64*2);
  _Float16* W3h  = (_Float16*)carve(256*128*2);
  float*    st   = (float*)   carve(3*64*4);               // stats for 3 layers
  float* cA1 = (float*)carve(2*64*4);
  float* cB1 = (float*)carve(2*64*4);
  float* cA2 = (float*)carve(2*128*4);
  float* cB2 = (float*)carve(2*128*4);
  float* cA3 = (float*)carve(2*256*4);
  float* cB3 = (float*)carve(2*256*4);
  float* mv  = (float*)carve(2*256*4);
  float* h1  = (float*)carve(2*512*4);
  float* h2  = (float*)carve(2*1024*4);

  k_zero<<<2048, 256, 0, stream>>>(acc, (size_t)2*SUBV*256);
  k_zero<<<1, 256, 0, stream>>>(st, 3*64);
  k_cvt_f16<<<(128*64  + 255)/256, 256, 0, stream>>>(w2, W2h, 128*64);
  k_cvt_f16<<<(256*128 + 255)/256, 256, 0, stream>>>(w3, W3h, 256*128);

  // layer 1 (4->64) + voxel indices
  k_layer1<<<NPTS/256, 256, 0, stream>>>(pc, w1, b1, bufA, vox);
  k_gnstats<64><<<dim3(512,2), 256, 0, stream>>>(bufA, st + 0);
  k_gnfin<64><<<1, 32, 0, stream>>>(st + 0, g1, be1, cA1, cB1);
  k_transform<64><<<(NPTS*64/8)/256, 256, 0, stream>>>(bufA, cA1, cB1, bufB);

  // layer 2 (64->128), WMMA, W staged in LDS via async copy
  k_gemm<64,128,2><<<NPTS/128, 256, 0, stream>>>(bufB, W2h, b2, bufA);
  k_gnstats<128><<<dim3(512,2), 256, 0, stream>>>(bufA, st + 64);
  k_gnfin<128><<<1, 32, 0, stream>>>(st + 64, g2, be2, cA2, cB2);
  k_transform<128><<<(NPTS*128/8)/256, 256, 0, stream>>>(bufA, cA2, cB2, bufB);

  // layer 3 (128->256), WMMA, W staged in LDS via async copy
  k_gemm<128,256,2><<<NPTS/128, 256, 0, stream>>>(bufB, W3h, b3, bufA);
  k_gnstats<256><<<dim3(512,2), 256, 0, stream>>>(bufA, st + 128);
  k_gnfin<256><<<1, 32, 0, stream>>>(st + 128, g3, be3, cA3, cB3);
  k_transform<256><<<(NPTS*256/8)/256, 256, 0, stream>>>(bufA, cA3, cB3, bufB);

  // voxel segment-sum + max
  k_scatter<<<NPTS*32/256, 256, 0, stream>>>(bufB, vox, acc);
  k_voxmax<<<dim3(8,2), 1024, 0, stream>>>(acc, mv);

  // head MLP
  k_head<256,512,true ><<<2, 256, 0, stream>>>(mv, lw1, lb1, lg1, lbe1, h1);
  k_head<512,1024,true><<<2, 256, 0, stream>>>(h1, lw2, lb2, lg2, lbe2, h2);
  k_head<1024,2048,false><<<2, 256, 0, stream>>>(h2, lw3, lb3, nullptr, nullptr,
                                                 (float*)d_out);
}
